// GCNGrasp_7456063226439
// MI455X (gfx1250) — matrix operand order repaired and checked by reading, not compiled, not run on hardware
//
#include <hip/hip_runtime.h>

// GCN on MI455X (gfx1250, wave32). Per layer, one fused kernel per batch-copy b:
//   stage:   pre-swizzled bf16 weights -> LDS (cooperative, once per block)
//   Phase A: Hw_b = H_b[156xK] * W[Kx128] via v_wmma_f32_16x16x32_bf16,
//            accumulators packed straight into LDS in WMMA B-fragment layout.
//   Phase B: Out_b = Adj~[160x160 bf16 dense] * Hw_b (LDS) + fused BN/ReLU.
// Densified normalized adjacency turns scatter/segment-sum into a tiny-K GEMM.
// Single 82MB activation buffer, in-place across layers -> L2 resident.

#define V_NODES 156
#define VPAD    160
#define CH      128

typedef __attribute__((ext_vector_type(16))) __bf16 v16bf;
typedef __attribute__((ext_vector_type(2)))  __bf16 v2bf;
typedef __attribute__((ext_vector_type(8)))  float  v8f;

union Frag16 { v16bf v; unsigned u[8]; };

#if defined(__has_builtin)
#if __has_builtin(__builtin_amdgcn_cvt_pk_bf16_f32)
#define HAVE_CVT_PK_BF16 1
#endif
#endif

// f32 pair -> packed bf16 dword (lo = a, hi = b), round-to-nearest-even
__device__ __forceinline__ unsigned pack_bf16(float a, float b) {
#ifdef HAVE_CVT_PK_BF16
  auto r = __builtin_amdgcn_cvt_pk_bf16_f32(a, b);
  return __builtin_bit_cast(unsigned, r);
#else
  // scalar fptrunc is RNE; backend picks native v_cvt when available
  v2bf t;
  t[0] = (__bf16)a;
  t[1] = (__bf16)b;
  return __builtin_bit_cast(unsigned, t);
#endif
}

// ---------------- setup kernels (tiny) ----------------

__global__ void prep_zero(float* A32, float* deg) {
  int t = blockIdx.x * blockDim.x + threadIdx.x;
  if (t < VPAD * VPAD) A32[t] = 0.f;
  if (t < VPAD) deg[t] = 0.f;
}

__global__ void prep_deg(const int* ei, int E, float* deg) {
  int t = blockIdx.x * blockDim.x + threadIdx.x;
  if (t < E) atomicAdd(&deg[ei[E + t]], 1.f);            // dst in-degree
  else if (t < E + V_NODES) atomicAdd(&deg[t - E], 1.f); // self loop
}

__global__ void prep_dinv(const float* deg, float* dinv) {
  int v = threadIdx.x;
  if (v < VPAD) dinv[v] = (v < V_NODES && deg[v] > 0.f) ? rsqrtf(deg[v]) : 0.f;
}

__global__ void prep_fillA(const int* ei, int E, const float* dinv, float* A32) {
  int t = blockIdx.x * blockDim.x + threadIdx.x;
  if (t < E) {
    int s = ei[t], d = ei[E + t];
    atomicAdd(&A32[d * VPAD + s], dinv[s] * dinv[d]);
  } else if (t < E + V_NODES) {
    int v = t - E;
    atomicAdd(&A32[v * VPAD + v], dinv[v] * dinv[v]);
  }
}

__global__ void prep_cvtA(const float* A32, unsigned* A16) {
  int t = blockIdx.x * blockDim.x + threadIdx.x;
  if (t < VPAD * VPAD / 2) A16[t] = pack_bf16(A32[2 * t], A32[2 * t + 1]);
}

// fold bias + inference BatchNorm into per-channel scale/shift (6 layers)
__global__ void prep_scale(const float* b0, const float* bh, const float* gamma,
                           const float* beta, const float* mean, const float* var,
                           float* scl, float* sft) {
  int t = blockIdx.x * blockDim.x + threadIdx.x;
  if (t >= 6 * CH) return;
  int l = t >> 7, c = t & 127;
  if (l < 5) {
    float sc = gamma[l * CH + c] * rsqrtf(var[l * CH + c] + 1e-5f);
    float bias = (l == 0) ? b0[c] : bh[(l - 1) * CH + c];
    scl[t] = sc;
    sft[t] = (bias - mean[l * CH + c]) * sc + beta[l * CH + c];
  } else {
    scl[t] = 1.f;
    sft[t] = bh[4 * CH + c];
  }
}

// swizzle weights [K][128] f32 -> bf16 B-fragment layout:
// dword index = ((kt*8+nt)*32+lane)*8+p ; k0 = kt*32 + (lane>>4)*8 + (p<4 ? 2p : 8+2p) ;
// n = nt*16 + (lane&15)   (matches A table order: half h holds k 8h..8h+7, 16+8h..16+8h+7)
__global__ void prep_wsw(const float* W, unsigned* Wsw, int K, int total) {
  int t = blockIdx.x * blockDim.x + threadIdx.x;
  if (t >= total) return;
  int p = t & 7, lane = (t >> 3) & 31, nt = (t >> 8) & 7, kt = t >> 11;
  int k0 = kt * 32 + (lane >> 4) * 8 + (p < 4 ? 2 * p : 8 + 2 * p);
  int d  = nt * 16 + (lane & 15);
  float f0 = (k0     < K) ? W[(long)k0 * CH + d]       : 0.f;
  float f1 = (k0 + 1 < K) ? W[(long)(k0 + 1) * CH + d] : 0.f;
  Wsw[t] = pack_bf16(f0, f1);
}

// ---------------- fused GCN layer: one block per batch element b ----------------
// 320 threads = 10 waves; wave w owns M-tile vt=w (16 rows) in both phases.
template <int K, bool RELU>
__global__ __launch_bounds__(320) void gcn_layer(
    const float* Hin, const unsigned* __restrict__ Wsw,
    const unsigned* __restrict__ A16,
    const float* __restrict__ scl, const float* __restrict__ sft,
    float* Hout) {
  constexpr int KT = (K + 31) / 32;
  // [0,10240): Hw_b 160x128 bf16 in B-frag layout; [10240,+KT*2048): weights
  __shared__ __align__(16) unsigned smem[10240 + KT * 2048];
  unsigned* ldsB = smem;
  unsigned* ldsW = smem + 10240;

  const int b    = blockIdx.x;
  const int tid  = threadIdx.x;
  const int lane = tid & 31;
  const int vt   = tid >> 5;       // wave id == M-tile id (0..9)
  const int half = lane >> 4;
  const int l15  = lane & 15;
  const float* Hb = Hin + (long)b * V_NODES * K;

  // cooperative stage: pre-swizzled weights -> LDS (b128 copies)
  {
    const uint4* src = (const uint4*)Wsw;
    uint4* dst = (uint4*)ldsW;
    for (int i = tid; i < KT * 512; i += 320) dst[i] = src[i];
  }
  __syncthreads();

  // ---- Phase A: rows [vt*16, vt*16+16) of Hw_b = H_b * W ----
  {
    const int u = vt * 16 + l15;
    const bool rowok = (u < V_NODES);
    const float* arow = Hb + (long)u * K;
    v8f acc[8] = {};
#pragma unroll
    for (int kt = 0; kt < KT; ++kt) {
      // A-fragment (ISA 7.12.2 16-bit A): half 0: K kb..kb+7, kb+16..kb+23
      Frag16 a;
      const int kb = kt * 32 + half * 8;
      if (rowok && (kb + 24 <= K)) {
#pragma unroll
        for (int p = 0; p < 4; ++p)
          a.u[p] = pack_bf16(arow[kb + 2 * p], arow[kb + 2 * p + 1]);
#pragma unroll
        for (int p = 0; p < 4; ++p)
          a.u[4 + p] = pack_bf16(arow[kb + 16 + 2 * p], arow[kb + 17 + 2 * p]);
      } else {  // row padding (u>=156) or K tail (layer 0: K=301)
#pragma unroll
        for (int p = 0; p < 8; ++p) {
          int k0 = kb + (p < 4 ? 2 * p : 8 + 2 * p);
          float f0 = (rowok && k0     < K) ? arow[k0]     : 0.f;
          float f1 = (rowok && k0 + 1 < K) ? arow[k0 + 1] : 0.f;
          a.u[p] = pack_bf16(f0, f1);
        }
      }
      // weight B-fragments come from LDS (low latency, shared by 10 waves)
      Frag16 bf[8];
#pragma unroll
      for (int nt = 0; nt < 8; ++nt)
        bf[nt].v = *(const v16bf*)&ldsW[((kt * 8 + nt) * 32 + lane) << 3];
#pragma unroll
      for (int nt = 0; nt < 8; ++nt)
        acc[nt] = __builtin_amdgcn_wmma_f32_16x16x32_bf16(
            false, a.v, false, bf[nt].v, (short)0, acc[nt], false, false);
    }
    // D-fragment rows j,j+1 live in adjacent VGPRs of the same lane ->
    // pack row-pairs straight into GEMM2 B-fragment dwords in LDS.
#pragma unroll
    for (int nt = 0; nt < 8; ++nt) {
#pragma unroll
      for (int t = 0; t < 4; ++t) {
        unsigned wv = pack_bf16(acc[nt][2 * t], acc[nt][2 * t + 1]);
        int r   = vt * 8 + half * 4 + t;      // k-pair index 0..79
        int kt2 = r >> 4, rp = r & 15;
        int hf  = rp >> 3, p = rp & 7;
        ldsB[(((kt2 * 8 + nt) * 32 + (l15 + (hf << 4))) << 3) + p] = wv;
      }
    }
  }

  // preload adjacency A-fragments (global, L1-hot) behind the barrier wait
  Frag16 aAdj[5];
#pragma unroll
  for (int kt = 0; kt < 5; ++kt) {
    const unsigned* ar =
        A16 + (((vt * 16 + l15) * VPAD + kt * 32 + half * 16) >> 1);
#pragma unroll
    for (int p = 0; p < 4; ++p) aAdj[kt].u[p] = ar[p];
#pragma unroll
    for (int p = 0; p < 4; ++p) aAdj[kt].u[4 + p] = ar[4 + p];
  }
  __syncthreads();

  // ---- Phase B: Out rows [vt*16, vt*16+16) = Adj~ * Hw_b (LDS), + BN/ReLU ----
  {
    v8f acc[8] = {};
#pragma unroll
    for (int kt = 0; kt < 5; ++kt) {
      Frag16 bf[8];
#pragma unroll
      for (int nt = 0; nt < 8; ++nt)
        bf[nt].v = *(const v16bf*)&ldsB[((kt * 8 + nt) * 32 + lane) << 3];
#pragma unroll
      for (int nt = 0; nt < 8; ++nt)
        acc[nt] = __builtin_amdgcn_wmma_f32_16x16x32_bf16(
            false, aAdj[kt].v, false, bf[nt].v, (short)0, acc[nt], false, false);
    }
#pragma unroll
    for (int nt = 0; nt < 8; ++nt) {
      int dc = nt * 16 + l15;
      float sc = scl[dc], sh = sft[dc];
#pragma unroll
      for (int j = 0; j < 8; ++j) {
        int v = vt * 16 + j + half * 8;
        if (v < V_NODES) {
          float o = acc[nt][j] * sc + sh;
          if constexpr (RELU) o = fmaxf(o, 0.f);
          Hout[((long)b * V_NODES + v) * CH + dc] = o;
        }
      }
    }
  }
}

extern "C" void kernel_launch(void* const* d_in, const int* in_sizes, int n_in,
                              void* d_out, int out_size, void* d_ws, size_t ws_size,
                              hipStream_t stream) {
  const float* x     = (const float*)d_in[0];
  const int*   ei    = (const int*)d_in[1];
  const float* W0    = (const float*)d_in[2];
  const float* b0    = (const float*)d_in[3];
  const float* Wh    = (const float*)d_in[4];
  const float* bh    = (const float*)d_in[5];
  const float* gamma = (const float*)d_in[6];
  const float* beta  = (const float*)d_in[7];
  const float* mean  = (const float*)d_in[8];
  const float* var   = (const float*)d_in[9];

  const int E   = in_sizes[1] / 2;
  const int CIN = 301;                          // fixed by the model
  const int Bsz = out_size / (V_NODES * CH);    // 1024
  const int KT0 = (CIN + 31) / 32;              // 10

  // workspace carve-up
  char* ws = (char*)d_ws;
  size_t off = 0;
  unsigned* Wsw0 = (unsigned*)(ws + off); off += (size_t)KT0 * 2048 * 4;
  unsigned* WswH = (unsigned*)(ws + off); off += (size_t)5 * 8192 * 4;
  float*    A32  = (float*)(ws + off);    off += (size_t)VPAD * VPAD * 4;
  unsigned* A16  = (unsigned*)(ws + off); off += (size_t)VPAD * VPAD / 2 * 4;
  float*    deg  = (float*)(ws + off);    off += VPAD * 4;
  float*    dinv = (float*)(ws + off);    off += VPAD * 4;
  float*    scl  = (float*)(ws + off);    off += 6 * CH * 4;
  float*    sft  = (float*)(ws + off);    off += 6 * CH * 4;
  off = (off + 255) & ~(size_t)255;
  float*    Buf  = (float*)(ws + off);    // Bsz*156*128 f32 (~82MB), in-place

  // graph + parameter prep
  prep_zero<<<(VPAD * VPAD + 255) / 256, 256, 0, stream>>>(A32, deg);
  prep_deg<<<(E + V_NODES + 255) / 256, 256, 0, stream>>>(ei, E, deg);
  prep_dinv<<<1, VPAD, 0, stream>>>(deg, dinv);
  prep_fillA<<<(E + V_NODES + 255) / 256, 256, 0, stream>>>(ei, E, dinv, A32);
  prep_cvtA<<<(VPAD * VPAD / 2 + 255) / 256, 256, 0, stream>>>(A32, A16);
  prep_scale<<<3, 256, 0, stream>>>(b0, bh, gamma, beta, mean, var, scl, sft);
  prep_wsw<<<(KT0 * 2048 + 255) / 256, 256, 0, stream>>>(W0, Wsw0, CIN, KT0 * 2048);
  for (int l = 0; l < 5; ++l)
    prep_wsw<<<(4 * 2048 + 255) / 256, 256, 0, stream>>>(
        Wh + (size_t)l * CH * CH, WswH + (size_t)l * 8192, CH, 4 * 2048);

  // 6 fused layers, one block per batch element, in-place activation buffer
  gcn_layer<301, true><<<Bsz, 320, 0, stream>>>(x, Wsw0, A16, scl, sft, Buf);
  for (int l = 1; l <= 4; ++l)
    gcn_layer<128, true><<<Bsz, 320, 0, stream>>>(
        Buf, WswH + (size_t)(l - 1) * 8192, A16, scl + l * CH, sft + l * CH, Buf);
  gcn_layer<128, false><<<Bsz, 320, 0, stream>>>(
      Buf, WswH + (size_t)4 * 8192, A16, scl + 5 * CH, sft + 5 * CH,
      (float*)d_out);
}
